// CRPClassifier_62869731279009
// MI455X (gfx1250) — compile-verified, matrix operand order checked by reading, not run on hardware
//
#include <hip/hip_runtime.h>

// ---------------------------------------------------------------------------
// CRP classifier forward for MI455X (gfx1250, wave32, WMMA bf16 path).
// Shapes: B=512, I=1024, HD=2048, L=1000, T_MAX=32.
// Defaults: KAPPA=1 -> rho=C=0.95, denom=0.05, rho/denom=19.
// Compute-bound (~207 GFLOP, weights L2-resident) -> bf16 WMMA w/ f32 acc,
// software-pipelined register double-buffering in the K loop.
// ---------------------------------------------------------------------------

#define Bsz 512
#define Isz 1024
#define HD  2048
#define Lsz 1000
#define LP  1024    // padded L
#define TMAX 32

typedef __attribute__((ext_vector_type(16))) __bf16 v16bf;
typedef __attribute__((ext_vector_type(8)))  float  v8f;
typedef __attribute__((ext_vector_type(4)))  unsigned int u32x4;

// ---- fp32 -> bf16 (round to nearest even) ---------------------------------
__device__ __forceinline__ unsigned short f2bf(float f) {
    unsigned int u = __float_as_uint(f);
    u += 0x7FFFu + ((u >> 16) & 1u);
    return (unsigned short)(u >> 16);
}

// ---- WMMA fragment load: 16 bf16 per lane as two 16B contiguous chunks ----
// A (16x32, row-major): lane = 16*kh + m; elems 0..7 -> kh*8+e,
// elems 8..15 -> 16+kh*8+(e-8).  B loaded from transposed [N][K] storage
// with the symmetric layout (col striped across lanes).  `p` already
// includes the kh*8 offset.
struct alignas(32) FragBits { u32x4 lo, hi; };

__device__ __forceinline__ v16bf load_frag(const unsigned short* __restrict__ p) {
    FragBits fb;
    fb.lo = *reinterpret_cast<const u32x4*>(p);
    fb.hi = *reinterpret_cast<const u32x4*>(p + 16);
    v16bf v;
    __builtin_memcpy(&v, &fb, sizeof(v));
    return v;
}

// ---------------------------------------------------------------------------
// Generic bf16 WMMA GEMM:  C[M,N] = A[M,K](bf16) * BT[N,K](bf16)^T  + epilogue
// Block = 128 threads = 4 waves; block tile 64(M) x 64(N); each wave owns a
// 16x64 strip (4 C fragments, A fragment reused 4x per K-step of 32).
// K loop is register double-buffered: loads for step k+32 are issued before
// the four WMMAs of step k, so the wave never stalls on L2 latency.
// MODE 0: Cout = acc + bias[n]                       (drive = x@W_IH + BH)
// MODE 1: recurrence step epilogue (LeakyReLU, active gating, dH inf-norm)
// MODE 2: Cout = acc + bias[n] for n < NL            (logits = H@W_HL + BL)
// ---------------------------------------------------------------------------
template <int MODE>
__global__ __launch_bounds__(128)
void wmma_gemm(const unsigned short* __restrict__ A,
               const unsigned short* __restrict__ BT,
               int M, int N, int K,
               const float* __restrict__ bias,
               const float* __restrict__ drive,
               const float* __restrict__ Hold,
               const int*   __restrict__ active,
               float* __restrict__ Cout,
               unsigned short* __restrict__ HbOut,
               unsigned int* __restrict__ dHbits,
               int NL) {
    const int lane = threadIdx.x & 31;
    const int wave = threadIdx.x >> 5;
    const int rowBase = blockIdx.y * 64 + wave * 16;
    const int colBase = blockIdx.x * 64;
    const int r16 = lane & 15;
    const int kh  = lane >> 4;

    // Per-lane streaming pointers (include the kh*8 element offset).
    const unsigned short* pa = A + (size_t)(rowBase + r16) * K + kh * 8;
    const unsigned short* pb0 = BT + (size_t)(colBase + r16) * K + kh * 8;
    const unsigned short* pb1 = pb0 + (size_t)16 * K;
    const unsigned short* pb2 = pb0 + (size_t)32 * K;
    const unsigned short* pb3 = pb0 + (size_t)48 * K;

    v8f acc0 = {0.f,0.f,0.f,0.f,0.f,0.f,0.f,0.f};
    v8f acc[4];
    acc[0] = acc0; acc[1] = acc0; acc[2] = acc0; acc[3] = acc0;

    // ---- prologue: load fragments for k0 = 0 ----
    v16bf a_cur = load_frag(pa);
    v16bf b_cur0 = load_frag(pb0);
    v16bf b_cur1 = load_frag(pb1);
    v16bf b_cur2 = load_frag(pb2);
    v16bf b_cur3 = load_frag(pb3);

    // ---- steady state: issue next-step loads, then compute current ----
    for (int k0 = 0; k0 < K - 32; k0 += 32) {
        pa += 32; pb0 += 32; pb1 += 32; pb2 += 32; pb3 += 32;
        const v16bf a_nxt  = load_frag(pa);
        const v16bf b_nxt0 = load_frag(pb0);
        const v16bf b_nxt1 = load_frag(pb1);
        const v16bf b_nxt2 = load_frag(pb2);
        const v16bf b_nxt3 = load_frag(pb3);
        __builtin_prefetch(pa + 32, 0, 3);   // near-cache speculative prefetch

        acc[0] = __builtin_amdgcn_wmma_f32_16x16x32_bf16(
            false, a_cur, false, b_cur0, (short)0, acc[0], false, false);
        acc[1] = __builtin_amdgcn_wmma_f32_16x16x32_bf16(
            false, a_cur, false, b_cur1, (short)0, acc[1], false, false);
        acc[2] = __builtin_amdgcn_wmma_f32_16x16x32_bf16(
            false, a_cur, false, b_cur2, (short)0, acc[2], false, false);
        acc[3] = __builtin_amdgcn_wmma_f32_16x16x32_bf16(
            false, a_cur, false, b_cur3, (short)0, acc[3], false, false);

        a_cur = a_nxt;
        b_cur0 = b_nxt0; b_cur1 = b_nxt1; b_cur2 = b_nxt2; b_cur3 = b_nxt3;
    }
    // ---- epilogue K-step ----
    acc[0] = __builtin_amdgcn_wmma_f32_16x16x32_bf16(
        false, a_cur, false, b_cur0, (short)0, acc[0], false, false);
    acc[1] = __builtin_amdgcn_wmma_f32_16x16x32_bf16(
        false, a_cur, false, b_cur1, (short)0, acc[1], false, false);
    acc[2] = __builtin_amdgcn_wmma_f32_16x16x32_bf16(
        false, a_cur, false, b_cur2, (short)0, acc[2], false, false);
    acc[3] = __builtin_amdgcn_wmma_f32_16x16x32_bf16(
        false, a_cur, false, b_cur3, (short)0, acc[3], false, false);

    // --- output epilogue ----------------------------------------------------
    bool rowActive[8];
    float dmax[8];
    if (MODE == 1) {
#pragma unroll
        for (int r = 0; r < 8; ++r) {
            rowActive[r] = active[rowBase + r + 8 * kh] != 0;
            dmax[r] = 0.f;
        }
    }

#pragma unroll
    for (int f = 0; f < 4; ++f) {
        const int n = colBase + f * 16 + r16;
#pragma unroll
        for (int r = 0; r < 8; ++r) {
            const int m = rowBase + r + 8 * kh;
            const float v = acc[f][r];
            if (MODE == 0) {
                Cout[(size_t)m * N + n] = v + bias[n];
            } else if (MODE == 2) {
                if (n < NL) Cout[(size_t)m * N + n] = v + bias[n];
            } else {  // MODE == 1: pre -> LeakyReLU -> gated H update
                const size_t idx = (size_t)m * N + n;
                const float pre = v + drive[idx];
                const float actv = pre >= 0.f ? pre : 0.05f * pre;
                const float ho = Hold[idx];
                const float hn = rowActive[r] ? actv : ho;
                Cout[idx] = hn;
                HbOut[idx] = f2bf(hn);
                dmax[r] = fmaxf(dmax[r], fabsf(hn - ho));
            }
        }
    }

    if (MODE == 1) {
#pragma unroll
        for (int r = 0; r < 8; ++r) {
            if (dmax[r] > 0.f) {  // |x| bits are monotone for non-negative f32
                atomicMax(dHbits + (rowBase + r + 8 * kh), __float_as_uint(dmax[r]));
            }
        }
    }
}

// ---------------------------------------------------------------------------
// Column L1 sums: colsum[j] = sum_i |A[i,j]*Msk[i,j]|  (one block per column)
// ---------------------------------------------------------------------------
__global__ __launch_bounds__(256)
void col_l1_kernel(const float* __restrict__ A, const float* __restrict__ Msk,
                   float* __restrict__ colsum, int rows, int cols) {
    __shared__ float sm[256];
    const int j = blockIdx.x;
    float s = 0.f;
    for (int i = threadIdx.x; i < rows; i += 256) {
        const size_t idx = (size_t)i * cols + j;
        s += fabsf(A[idx] * Msk[idx]);
    }
    sm[threadIdx.x] = s;
    __syncthreads();
    for (int st = 128; st > 0; st >>= 1) {
        if (threadIdx.x < st) sm[threadIdx.x] += sm[threadIdx.x + st];
        __syncthreads();
    }
    if (threadIdx.x == 0) colsum[j] = sm[0];
}

// scalars[0] = scale = 0.95/max(max_j col_l1, 1e-8); scalars[1] = gamma_coef
__global__ __launch_bounds__(256)
void scalars_kernel(const float* __restrict__ col_l1,
                    const float* __restrict__ whl,
                    float* __restrict__ scalars) {
    __shared__ float sm[256];
    const int t = threadIdx.x;
    float mx = -1e30f;
    for (int j = t; j < HD; j += 256) mx = fmaxf(mx, col_l1[j]);
    sm[t] = mx; __syncthreads();
    for (int s = 128; s > 0; s >>= 1) {
        if (t < s) sm[t] = fmaxf(sm[t], sm[t + s]);
        __syncthreads();
    }
    const float m1 = sm[0];
    __syncthreads();
    mx = -1e30f;
    for (int j = t; j < Lsz; j += 256) mx = fmaxf(mx, whl[j]);
    sm[t] = mx; __syncthreads();
    for (int s = 128; s > 0; s >>= 1) {
        if (t < s) sm[t] = fmaxf(sm[t], sm[t + s]);
        __syncthreads();
    }
    if (t == 0) {
        scalars[0] = 0.95f / fmaxf(m1, 1e-8f);
        scalars[1] = sm[0] * 19.0f;   // whl_inf * rho/(1-rho)
    }
}

// WT[n*K + k] = bf16(R[k,n] * M[k,n] * (sel==0 ? scale : 1)), zero for n>=NL
__global__ __launch_bounds__(256)
void build_wt_kernel(const float* __restrict__ R, const float* __restrict__ M,
                     const float* __restrict__ scalars, int sel,
                     unsigned short* __restrict__ WT,
                     int K, int N, int srcld, int NL) {
    const size_t idx = (size_t)blockIdx.x * 256 + threadIdx.x;
    const size_t total = (size_t)N * K;
    if (idx >= total) return;
    const int n = (int)(idx / K);
    const int k = (int)(idx % K);
    float v = 0.f;
    if (n < NL) {
        const size_t s = (size_t)k * srcld + n;
        v = R[s] * M[s];
        if (sel == 0) v *= scalars[0];
    }
    WT[idx] = f2bf(v);
}

__global__ __launch_bounds__(256)
void to_bf16_kernel(const float* __restrict__ src, unsigned short* __restrict__ dst,
                    int n) {
    const int i = blockIdx.x * 256 + threadIdx.x;
    if (i < n) dst[i] = f2bf(src[i]);
}

__global__ __launch_bounds__(256)
void init_state(float* __restrict__ H0, unsigned short* __restrict__ Hb0,
                unsigned int* __restrict__ dH, int* __restrict__ active) {
    const size_t n = (size_t)Bsz * HD;
    for (size_t t = (size_t)blockIdx.x * 256 + threadIdx.x; t < n;
         t += (size_t)gridDim.x * 256) {
        H0[t] = 0.f;
        Hb0[t] = 0;
    }
    const int i = blockIdx.x * 256 + threadIdx.x;
    if (i < Bsz) { dH[i] = 0u; active[i] = 1; }
}

// ---------------------------------------------------------------------------
// Per-row certification: wave32 top-2 over 1000 logits, margin vs 2*Gamma.
// One wave per row.  On last step, never-certified rows flush too.
// ---------------------------------------------------------------------------
__global__ __launch_bounds__(256)
void update_rows(const float* __restrict__ logits,   // [B, LP]
                 const float* __restrict__ scalars,  // [1] = gamma_coef
                 unsigned int* __restrict__ dHbits,
                 int* __restrict__ active,
                 float* __restrict__ out,            // [B, L]
                 int is_last) {
    const int lane = threadIdx.x & 31;
    const int wave = threadIdx.x >> 5;
    const int row = blockIdx.x * 8 + wave;
    if (row >= Bsz) return;

    if (active[row] != 0) {
        const float* lr = logits + (size_t)row * LP;
        float m1 = -1e30f, m2 = -1e30f;
        for (int n = lane; n < Lsz; n += 32) {
            const float v = lr[n];
            if (v > m1) { m2 = m1; m1 = v; }
            else if (v > m2) { m2 = v; }
        }
        for (int off = 16; off > 0; off >>= 1) {
            const float o1 = __shfl_xor(m1, off, 32);
            const float o2 = __shfl_xor(m2, off, 32);
            const float hi = fmaxf(m1, o1);
            const float lo = fmaxf(fminf(m1, o1), fmaxf(m2, o2));
            m1 = hi; m2 = lo;
        }
        const float margin = m1 - m2;
        const float dh = __uint_as_float(dHbits[row]);
        const float Gamma = scalars[1] * dh;
        const bool newly = margin > 2.0f * Gamma;
        if (newly || is_last) {
            for (int n = lane; n < Lsz; n += 32)
                out[(size_t)row * Lsz + n] = lr[n];
            if (lane == 0) active[row] = 0;
        }
    }
    if (lane == 0) dHbits[row] = 0u;   // reset for next step
}

// ---------------------------------------------------------------------------
static inline char* wsalloc(char*& p, size_t bytes) {
    char* r = p;
    p += (bytes + 255) & ~(size_t)255;
    return r;
}

extern "C" void kernel_launch(void* const* d_in, const int* in_sizes, int n_in,
                              void* d_out, int out_size, void* d_ws, size_t ws_size,
                              hipStream_t stream) {
    (void)in_sizes; (void)n_in; (void)out_size; (void)ws_size;
    const float* x   = (const float*)d_in[0];
    const float* RIH = (const float*)d_in[1];
    const float* RH  = (const float*)d_in[2];
    const float* RHL = (const float*)d_in[3];
    const float* BH  = (const float*)d_in[4];
    const float* BL  = (const float*)d_in[5];
    const float* MIH = (const float*)d_in[6];
    const float* MH  = (const float*)d_in[7];
    const float* MHL = (const float*)d_in[8];
    float* out = (float*)d_out;

    char* p = (char*)d_ws;
    unsigned short* WT_H  = (unsigned short*)wsalloc(p, (size_t)HD * HD * 2);
    unsigned short* WT_HL = (unsigned short*)wsalloc(p, (size_t)LP * HD * 2);
    unsigned short* WT_IH = (unsigned short*)wsalloc(p, (size_t)HD * Isz * 2);
    unsigned short* xb    = (unsigned short*)wsalloc(p, (size_t)Bsz * Isz * 2);
    float* drive = (float*)wsalloc(p, (size_t)Bsz * HD * 4);
    float* Hbuf0 = (float*)wsalloc(p, (size_t)Bsz * HD * 4);
    float* Hbuf1 = (float*)wsalloc(p, (size_t)Bsz * HD * 4);
    unsigned short* Hb0 = (unsigned short*)wsalloc(p, (size_t)Bsz * HD * 2);
    unsigned short* Hb1 = (unsigned short*)wsalloc(p, (size_t)Bsz * HD * 2);
    float* logits = (float*)wsalloc(p, (size_t)Bsz * LP * 4);
    float* col_l1 = (float*)wsalloc(p, HD * 4);
    float* whl    = (float*)wsalloc(p, LP * 4);
    float* scalars = (float*)wsalloc(p, 256);
    unsigned int* dHb = (unsigned int*)wsalloc(p, Bsz * 4);
    int* active = (int*)wsalloc(p, Bsz * 4);

    // ---- preprocessing ----
    col_l1_kernel<<<HD, 256, 0, stream>>>(RH, MH, col_l1, HD, HD);
    col_l1_kernel<<<Lsz, 256, 0, stream>>>(RHL, MHL, whl, HD, Lsz);
    scalars_kernel<<<1, 256, 0, stream>>>(col_l1, whl, scalars);

    build_wt_kernel<<<(HD * HD) / 256, 256, 0, stream>>>(
        RH, MH, scalars, /*sel=*/0, WT_H, HD, HD, HD, HD);
    build_wt_kernel<<<(LP * HD) / 256, 256, 0, stream>>>(
        RHL, MHL, scalars, /*sel=*/1, WT_HL, HD, LP, Lsz, Lsz);
    build_wt_kernel<<<(HD * Isz) / 256, 256, 0, stream>>>(
        RIH, MIH, scalars, /*sel=*/1, WT_IH, Isz, HD, HD, HD);
    to_bf16_kernel<<<(Bsz * Isz) / 256, 256, 0, stream>>>(x, xb, Bsz * Isz);
    init_state<<<(Bsz * HD) / 256, 256, 0, stream>>>(Hbuf0, Hb0, dHb, active);

    // drive = x @ W_IH + BH   (M=512, N=2048, K=1024)
    wmma_gemm<0><<<dim3(HD / 64, Bsz / 64), 128, 0, stream>>>(
        xb, WT_IH, Bsz, HD, Isz, BH, nullptr, nullptr, nullptr,
        drive, nullptr, nullptr, HD);

    // ---- 32 recurrence steps, ping-pong H ----
    float* Hc[2] = {Hbuf0, Hbuf1};
    unsigned short* Hbc[2] = {Hb0, Hb1};
    for (int t = 1; t <= TMAX; ++t) {
        const int pc = (t - 1) & 1, pn = t & 1;
        // H_next = step(H) : pre = H@W_H + drive  (M=512, N=2048, K=2048)
        wmma_gemm<1><<<dim3(HD / 64, Bsz / 64), 128, 0, stream>>>(
            Hbc[pc], WT_H, Bsz, HD, HD, nullptr, drive, Hc[pc], active,
            Hc[pn], Hbc[pn], dHb, HD);
        // logits = H_next @ W_HL + BL  (M=512, N=1024 padded, K=2048)
        wmma_gemm<2><<<dim3(LP / 64, Bsz / 64), 128, 0, stream>>>(
            Hbc[pn], WT_HL, Bsz, LP, HD, BL, nullptr, nullptr, nullptr,
            logits, nullptr, nullptr, Lsz);
        // certification / output flush
        update_rows<<<Bsz / 8, 256, 0, stream>>>(
            logits, scalars, dHb, active, out, (t == TMAX) ? 1 : 0);
    }
}
